// PatchFSL_79645873537372
// MI455X (gfx1250) — matrix operand order repaired and checked by reading, not compiled, not run on hardware
//
#include <hip/hip_runtime.h>

#define N_WAY 5
#define K_SHOT 5
#define SEQ 196
#define SEQ_PAD 208
#define DIM 384
#define LS 4900        /* N_WAY*K_SHOT*SEQ */
#define LS_PAD 4912    /* 307*16 */
#define ROW_TILES 307
#define NSUP 25
#define NQRY 75
#define CLASS_ROWS 980 /* K_SHOT*SEQ */

typedef __attribute__((ext_vector_type(16))) _Float16 v16h;
typedef __attribute__((ext_vector_type(8)))  float    v8f;
typedef __attribute__((ext_vector_type(4)))  float    v4f;

// 1/TEMP * log2(e): exp(x/TEMP) == exp2(x * K_SCALE)
#define K_SCALE (1.4426950408889634f / 0.0510310363f)
#define K_COEF  (0.1f / (25.0f * 0.0510310363f))  /* LR/(B*TEMP) */

// ---------------------------------------------------------------------------
// Row L2-normalize (1e-8 floor) + split each fp32 value into f16 hi + f16 lo
// residual. One wave per (padded) output row. Pad rows are written as zeros.
// ---------------------------------------------------------------------------
__global__ void normalize_split_kernel(const float* __restrict__ src,
                                       unsigned short* __restrict__ hiO,
                                       unsigned short* __restrict__ loO,
                                       int paddedRows, int srcRows, int perImagePad) {
  int gtid = blockIdx.x * blockDim.x + threadIdx.x;
  int wave = gtid >> 5;
  int lane = gtid & 31;
  if (wave >= paddedRows) return;

  int srcRow;
  bool valid;
  if (perImagePad) {
    int b = wave / SEQ_PAD;
    int s = wave - b * SEQ_PAD;
    valid = (s < SEQ);
    srcRow = b * SEQ + s;
  } else {
    valid = (wave < srcRows);
    srcRow = wave;
  }

  float x[12];
  float ss = 0.f;
  if (valid) {
    const float* r = src + (size_t)srcRow * DIM;
#pragma unroll
    for (int j = 0; j < 12; ++j) { x[j] = r[lane + 32 * j]; ss += x[j] * x[j]; }
  } else {
#pragma unroll
    for (int j = 0; j < 12; ++j) x[j] = 0.f;
  }
#pragma unroll
  for (int m = 16; m >= 1; m >>= 1) ss += __shfl_xor(ss, m, 32);

  float inv = valid ? (1.f / fmaxf(sqrtf(ss), 1e-8f)) : 0.f;
  _Float16* ho = (_Float16*)hiO + (size_t)wave * DIM;
  _Float16* lo = (_Float16*)loO + (size_t)wave * DIM;
#pragma unroll
  for (int j = 0; j < 12; ++j) {
    float y = x[j] * inv;
    _Float16 h = (_Float16)y;
    float res = y - (float)h;
    ho[lane + 32 * j] = h;
    lo[lane + 32 * j] = (_Float16)res;
  }
}

// ---------------------------------------------------------------------------
// Fused GEMM + exp + per-image row-sum:
//   T[row, img] = sum_{s<196} exp( dot(A_row, B_{img,s}) / TEMP )
// Split-f16: dot = Ahi*Bhi + Ahi*Blo + Alo*Bhi (f32 accum) ~ fp32 accuracy.
// One wave owns a 16-row tile x one image (13 col tiles of 16, K=384=12x32).
// maskDiag: zero contributions where row/196 == img (support-vs-support mask;
// exp((c-100)/TEMP) underflows to exactly 0 in fp32, so skipping is exact).
// ---------------------------------------------------------------------------
__global__ void __launch_bounds__(256) gemm_exp_rowsum_kernel(
    const unsigned short* __restrict__ AhiU, const unsigned short* __restrict__ AloU,
    const unsigned short* __restrict__ BhiU, const unsigned short* __restrict__ BloU,
    float* __restrict__ T, int nImg, int maskDiag) {
  const _Float16* Ahi = (const _Float16*)AhiU;
  const _Float16* Alo = (const _Float16*)AloU;
  const _Float16* Bhi = (const _Float16*)BhiU;
  const _Float16* Blo = (const _Float16*)BloU;

  int lane = threadIdx.x & 31;
  int half = lane >> 4;
  int l15  = lane & 15;
  int waveId = blockIdx.x * (blockDim.x >> 5) + (threadIdx.x >> 5);
  if (waveId >= ROW_TILES * nImg) return;  // whole-wave uniform exit (EXEC stays all-1 for WMMA)

  int rowTile = waveId % ROW_TILES;
  int img     = waveId / ROW_TILES;
  int rowBase = rowTile * 16;

  // A fragment (16x32, f16): lane l15 holds row rowBase+l15; half selects K phase.
  const _Float16* aBaseHi = Ahi + (size_t)(rowBase + l15) * DIM + 8 * half;
  const _Float16* aBaseLo = Alo + (size_t)(rowBase + l15) * DIM + 8 * half;

  float sums[8];
#pragma unroll
  for (int g = 0; g < 8; ++g) sums[g] = 0.f;

  for (int ct = 0; ct < 13; ++ct) {
    int colRow = img * SEQ_PAD + ct * 16 + l15;  // B column = padded query row
    const _Float16* bBaseHi = Bhi + (size_t)colRow * DIM + 8 * half;
    const _Float16* bBaseLo = Blo + (size_t)colRow * DIM + 8 * half;

    v8f acc = {};
    for (int ks = 0; ks < 12; ++ks) {
      union { v4f f[2]; v16h h; } ah, al, bh, bl;
      const _Float16* pa = aBaseHi + ks * 32;
      ah.f[0] = *(const v4f*)(pa);
      ah.f[1] = *(const v4f*)(pa + 16);
      const _Float16* pal = aBaseLo + ks * 32;
      al.f[0] = *(const v4f*)(pal);
      al.f[1] = *(const v4f*)(pal + 16);
      const _Float16* pb = bBaseHi + ks * 32;
      bh.f[0] = *(const v4f*)(pb);
      bh.f[1] = *(const v4f*)(pb + 16);
      const _Float16* pbl = bBaseLo + ks * 32;
      bl.f[0] = *(const v4f*)(pbl);
      bl.f[1] = *(const v4f*)(pbl + 16);

      acc = __builtin_amdgcn_wmma_f32_16x16x32_f16(false, ah.h, false, bh.h, (short)0, acc, false, false);
      acc = __builtin_amdgcn_wmma_f32_16x16x32_f16(false, ah.h, false, bl.h, (short)0, acc, false, false);
      acc = __builtin_amdgcn_wmma_f32_16x16x32_f16(false, al.h, false, bh.h, (short)0, acc, false, false);
    }

    int s = ct * 16 + l15;
    bool colValid = (s < SEQ);
#pragma unroll
    for (int g = 0; g < 8; ++g) {
      int row = rowBase + g + 8 * half;  // C layout: VGPR g holds M=g (lanes0-15) / M=g+8 (lanes16-31)
      bool masked = maskDiag && ((row / SEQ) == img);
      float e = (colValid && !masked) ? exp2f(acc[g] * K_SCALE) : 0.f;
      e += __shfl_xor(e, 1, 32);
      e += __shfl_xor(e, 2, 32);
      e += __shfl_xor(e, 4, 32);
      e += __shfl_xor(e, 8, 32);
      sums[g] += e;
    }
  }

  if (l15 == 0) {
#pragma unroll
    for (int g = 0; g < 8; ++g) {
      int row = rowBase + half * 8 + g;
      if (row < LS) T[(size_t)row * nImg + img] = sums[g];
    }
  }
}

// ---------------------------------------------------------------------------
// 15 SGD steps on v (starts at 0), entirely in one workgroup with v in LDS.
//   Z[w,b] = sum_{i in class w} u_i * Tss[i,b],  u_i = exp(v_i/TEMP)
//   v_i   += u_i * sum_b W[w_i,b]*Tss[i,b],  W = LR/(B*T)*(delta - softmax)/Z
// Emits u_i = exp(v_i/TEMP) for the final prediction.
// ---------------------------------------------------------------------------
__global__ void __launch_bounds__(1024) sgd_kernel(const float* __restrict__ Tss,
                                                   const int* __restrict__ labels,
                                                   float* __restrict__ uOut) {
  __shared__ float vlds[LS];
  __shared__ float part[32 * 128];
  __shared__ float Zs[128];
  __shared__ float Ws[128];

  int tid = threadIdx.x;
  int wave = tid >> 5, lane = tid & 31;
  for (int i = tid; i < LS; i += 1024) vlds[i] = 0.f;
  __syncthreads();

  for (int it = 0; it < 15; ++it) {
    // --- phase 1: per-wave partial Z, lanes 0..24 each own one b ---
    float accw[5] = {0.f, 0.f, 0.f, 0.f, 0.f};
    for (int i = wave; i < LS; i += 32) {
      float u = exp2f(vlds[i] * K_SCALE);
      int w = i / CLASS_ROWS;
      float val = (lane < 25) ? u * Tss[i * 25 + lane] : 0.f;
#pragma unroll
      for (int c = 0; c < 5; ++c) accw[c] += (w == c) ? val : 0.f;
    }
    if (lane < 25) {
#pragma unroll
      for (int c = 0; c < 5; ++c) part[wave * 128 + c * 25 + lane] = accw[c];
    }
    __syncthreads();
    if (tid < 125) {
      float z = 0.f;
      for (int wv = 0; wv < 32; ++wv) z += part[wv * 128 + tid];
      Zs[tid] = z;
    }
    __syncthreads();

    // --- phase 2: W coefficients, one thread per support image b ---
    if (tid < 25) {
      int b = tid;
      float p[5], m = -1e30f;
#pragma unroll
      for (int w = 0; w < 5; ++w) { p[w] = logf(Zs[w * 25 + b]); m = fmaxf(m, p[w]); }
      float se = 0.f;
#pragma unroll
      for (int w = 0; w < 5; ++w) { p[w] = expf(p[w] - m); se += p[w]; }
      int lb = labels[b];
#pragma unroll
      for (int w = 0; w < 5; ++w) {
        float sig = p[w] / se;
        float delta = (w == lb) ? 1.f : 0.f;
        Ws[w * 25 + b] = K_COEF * (delta - sig) / Zs[w * 25 + b];
      }
    }
    __syncthreads();

    // --- phase 3: update v ---
    for (int i = tid; i < LS; i += 1024) {
      float u = exp2f(vlds[i] * K_SCALE);
      int w = i / CLASS_ROWS;
      const float* wr = &Ws[w * 25];
      const float* tr = &Tss[i * 25];
      float s = 0.f;
#pragma unroll
      for (int b = 0; b < 25; ++b) s += wr[b] * tr[b];
      vlds[i] += u * s;
    }
    __syncthreads();
  }

  for (int i = tid; i < LS; i += 1024) uOut[i] = exp2f(vlds[i] * K_SCALE);
}

// ---------------------------------------------------------------------------
// out[b,w] = log( sum_{i in class w} u_i * Tsq[i,b] )   -> [75,5] row-major
// ---------------------------------------------------------------------------
__global__ void __launch_bounds__(256) final_kernel(const float* __restrict__ Tsq,
                                                    const float* __restrict__ u,
                                                    float* __restrict__ out) {
  int blk = blockIdx.x;         // 0..374 == b*5 + w
  int b = blk / 5, w = blk - 5 * b;
  __shared__ float red[256];
  float s = 0.f;
  int base = w * CLASS_ROWS;
  for (int k = threadIdx.x; k < CLASS_ROWS; k += 256) {
    int i = base + k;
    s += u[i] * Tsq[(size_t)i * NQRY + b];
  }
  red[threadIdx.x] = s;
  __syncthreads();
  for (int off = 128; off >= 1; off >>= 1) {
    if (threadIdx.x < off) red[threadIdx.x] += red[threadIdx.x + off];
    __syncthreads();
  }
  if (threadIdx.x == 0) out[blk] = logf(red[0]);
}

// ---------------------------------------------------------------------------
extern "C" void kernel_launch(void* const* d_in, const int* in_sizes, int n_in,
                              void* d_out, int out_size, void* d_ws, size_t ws_size,
                              hipStream_t stream) {
  (void)in_sizes; (void)n_in; (void)out_size; (void)ws_size;
  const float* supKey = (const float*)d_in[0];
  const float* supQry = (const float*)d_in[1];
  const float* qry    = (const float*)d_in[2];
  const int*   labels = (const int*)d_in[3];

  char* ws = (char*)d_ws;
  size_t off = 0;
  auto take = [&](size_t bytes) -> void* {
    void* p = ws + off;
    off += (bytes + 255) & ~(size_t)255;
    return p;
  };
  unsigned short* Khi  = (unsigned short*)take((size_t)LS_PAD * DIM * 2);
  unsigned short* Klo  = (unsigned short*)take((size_t)LS_PAD * DIM * 2);
  unsigned short* SQhi = (unsigned short*)take((size_t)NSUP * SEQ_PAD * DIM * 2);
  unsigned short* SQlo = (unsigned short*)take((size_t)NSUP * SEQ_PAD * DIM * 2);
  unsigned short* Qhi  = (unsigned short*)take((size_t)NQRY * SEQ_PAD * DIM * 2);
  unsigned short* Qlo  = (unsigned short*)take((size_t)NQRY * SEQ_PAD * DIM * 2);
  float* Tss = (float*)take((size_t)LS * NSUP * sizeof(float));
  float* Tsq = (float*)take((size_t)LS * NQRY * sizeof(float));
  float* uV  = (float*)take((size_t)LS * sizeof(float));

  // Normalize + f16 hi/lo split (one wave per padded row; 8 rows per block).
  {
    int rows = LS_PAD;
    normalize_split_kernel<<<(rows + 7) / 8, 256, 0, stream>>>(supKey, Khi, Klo, rows, LS, 0);
  }
  {
    int rows = NSUP * SEQ_PAD;
    normalize_split_kernel<<<(rows + 7) / 8, 256, 0, stream>>>(supQry, SQhi, SQlo, rows, 0, 1);
  }
  {
    int rows = NQRY * SEQ_PAD;
    normalize_split_kernel<<<(rows + 7) / 8, 256, 0, stream>>>(qry, Qhi, Qlo, rows, 0, 1);
  }

  // Fused GEMM + exp-rowsum: T_ss (with diag mask) and T_sq.
  {
    int waves = ROW_TILES * NSUP;
    gemm_exp_rowsum_kernel<<<(waves + 7) / 8, 256, 0, stream>>>(Khi, Klo, SQhi, SQlo, Tss, NSUP, 1);
  }
  {
    int waves = ROW_TILES * NQRY;
    gemm_exp_rowsum_kernel<<<(waves + 7) / 8, 256, 0, stream>>>(Khi, Klo, Qhi, Qlo, Tsq, NQRY, 0);
  }

  // 15-step SGD on v (single workgroup), then final logits.
  sgd_kernel<<<1, 1024, 0, stream>>>(Tss, labels, uV);
  final_kernel<<<375, 256, 0, stream>>>(Tsq, uV, (float*)d_out);
}
